// ProbTravelTime_spatial_25134148616286
// MI455X (gfx1250) — compile-verified, matrix operand order checked by reading, not run on hardware
//
#include <hip/hip_runtime.h>
#include <hip/hip_bf16.h>

typedef __attribute__((ext_vector_type(16))) __bf16        v16bf;
typedef __attribute__((ext_vector_type(8)))  float         v8f;
typedef __attribute__((ext_vector_type(8)))  unsigned int  v8u;

#define B_    512
#define S_    256
#define L_    64
#define DIMC  128
#define DIMR  256
#define HID_  512
#define CELLS 289   // 17*17

__device__ __forceinline__ unsigned int pack2bf(float lo, float hi) {
    unsigned int a = __builtin_bit_cast(unsigned int, lo);
    unsigned int b = __builtin_bit_cast(unsigned int, hi);
    a = (a + 0x7FFFu + ((a >> 16) & 1u)) >> 16;   // round-to-nearest-even bf16
    b = (b + 0x7FFFu + ((b >> 16) & 1u)) >> 16;
    return a | (b << 16);
}

// branch-free SELU: scale*max(x,0) + scale*alpha*(exp(min(x,0)) - 1)
//  x>0: exp(0)-1 = 0 -> scale*x ; x<=0: max term 0 -> scale*alpha*(exp(x)-1)
__device__ __forceinline__ float selu_f(float x) {
    const float scale  = 1.0507009873554805f;
    const float salpha = 1.7580993408473766f;   // scale * alpha
    return fmaf(scale, fmaxf(x, 0.f), salpha * (__expf(fminf(x, 0.f)) - 1.f));
}

// ---------------------------------------------------------------------------
// K0: swizzle f_W1[0:256,:] (f32, row-major 384x512) into bf16 WMMA B-fragments.
// Fragment (nt,kt): lane l<16 holds column nt*16+l with K {k0+0..7, k0+16..23};
// lane l+16 holds column nt*16+l with K {k0+8..15, k0+24..31}.
// Stored [frag][lane][2 x uint4] -> GEMM does two global b128 loads per lane.
// ---------------------------------------------------------------------------
__global__ void k_prep_w(const float* __restrict__ fW1, uint4* __restrict__ wfrag) {
    int frag = blockIdx.x;              // 0..255 = nt*8 + kt
    int nt = frag >> 3, kt = frag & 7;
    int lane = threadIdx.x;             // 0..31
    int l16 = lane & 15;
    int off = (lane < 16) ? 0 : 8;
    int n  = nt * 16 + l16;
    int k0 = kt * 32;
    unsigned int u[8];
#pragma unroll
    for (int p = 0; p < 4; ++p) {
        int k = k0 + off + 2 * p;
        u[p] = pack2bf(fW1[k * HID_ + n], fW1[(k + 1) * HID_ + n]);
    }
#pragma unroll
    for (int p = 0; p < 4; ++p) {
        int k = k0 + off + 16 + 2 * p;
        u[4 + p] = pack2bf(fW1[k * HID_ + n], fW1[(k + 1) * HID_ + n]);
    }
    uint4* dst = wfrag + (size_t)(frag * 32 + lane) * 2;
    dst[0] = make_uint4(u[0], u[1], u[2], u[3]);
    dst[1] = make_uint4(u[4], u[5], u[6], u[7]);
}

// ---------------------------------------------------------------------------
// K1+K2 fused: per-b gather-mean, f2 selu MLP, and per-b hidden bias
//   bias[b][n] = f_b1[n] + sum_k c_tr[b][k] * f_W1[256+k][n]
// ---------------------------------------------------------------------------
__global__ __launch_bounds__(256) void k_context(
        const float* __restrict__ c, const int* __restrict__ lon,
        const int* __restrict__ lat,
        const float* __restrict__ f2W1, const float* __restrict__ f2b1,
        const float* __restrict__ f2W2, const float* __restrict__ f2b2,
        const float* __restrict__ fW1,  const float* __restrict__ fb1,
        float* __restrict__ bias) {
    __shared__ int   sidx[L_];
    __shared__ float ctraj[DIMC];
    __shared__ float h[256];
    __shared__ float ctr[DIMC];
    int b = blockIdx.x, t = threadIdx.x;
    if (t < L_) sidx[t] = lat[b * L_ + t] * 17 + lon[b * L_ + t];
    __syncthreads();
    if (t < DIMC) {                      // gather-mean over trajectory cells
        float acc = 0.f;
        for (int l = 0; l < L_; ++l) acc += c[t * CELLS + sidx[l]];
        ctraj[t] = acc * (1.f / L_);
    }
    __syncthreads();
    {                                    // h = selu(c_traj @ f2_W1 + b1)
        float acc = f2b1[t];
        for (int k = 0; k < DIMC; ++k) acc += ctraj[k] * f2W1[k * 256 + t];
        h[t] = selu_f(acc);
    }
    __syncthreads();
    if (t < DIMC) {                      // c_tr = h @ f2_W2 + b2
        float acc = f2b2[t];
        for (int k = 0; k < 256; ++k) acc += h[k] * f2W2[k * DIMC + t];
        ctr[t] = acc;
    }
    __syncthreads();
#pragma unroll
    for (int j = 0; j < 2; ++j) {        // per-b hidden bias (512 wide)
        int n = t + j * 256;
        float acc = fb1[n];
        for (int k = 0; k < DIMC; ++k) acc += ctr[k] * fW1[(DIMR + k) * HID_ + n];
        bias[(size_t)b * HID_ + n] = acc;
    }
}

// ---------------------------------------------------------------------------
// K3: fused GEMM -> selu -> (W21,W22) dot. One block = one 16-row M-tile.
// rho is converted to bf16 during staging and laid out in LDS in WMMA
// fragment order, so A fragments are pure ds_load_b128 (no repack VALU).
// 4 waves split N=512 into 8 n-tiles each; 64 WMMAs/wave; h1 never hits memory.
// ---------------------------------------------------------------------------
__global__ __launch_bounds__(128) void k_gemm(
        const float* __restrict__ rho, const uint4* __restrict__ wfrag,
        const float* __restrict__ bias,
        const float* __restrict__ w21, const float* __restrict__ w22,
        const float* __restrict__ b21, const float* __restrict__ b22,
        float* __restrict__ logm, float* __restrict__ logv) {
    // A tile in bf16 fragment order: [kt(8)][lane(32)][half(16)] = 4096 halfs = 8 KB
    __shared__ unsigned short sA[8 * 32 * 16];
    __shared__ float redm[16];
    __shared__ float redv[16];
    int tid  = threadIdx.x;
    int wave = tid >> 5, lane = tid & 31;
    int tile = blockIdx.x;               // 0..8191
    int b    = tile >> 4;                // 16 tiles per batch element
    int s0   = (tile & 15) << 4;
    size_t row0 = (size_t)b * S_ + s0;
    if (tid < 16) { redm[tid] = 0.f; redv[tid] = 0.f; }

    // cooperative stage: read 16x256 f32 rho rows (coalesced float4), convert to
    // bf16, scatter into fragment-element order (one 8B LDS store per float4)
    {
        const float4* src = (const float4*)(rho + row0 * DIMR);
        for (int i = tid; i < 1024; i += 128) {
            int r = i >> 6, c4 = i & 63;             // row, float4 index
            float4 v = src[r * 64 + c4];
            int k    = c4 << 2;                      // k multiple of 4
            int kt   = k >> 5, koff = k & 31;
            int ln   = r + (((koff >> 3) & 1) << 4); // fragment lane
            int half = (koff & 7) + ((koff & 16) ? 8 : 0);
            unsigned int* d = (unsigned int*)&sA[kt * 512 + ln * 16 + half];
            d[0] = pack2bf(v.x, v.y);
            d[1] = pack2bf(v.z, v.w);
        }
    }
    __syncthreads();

    // A fragments: two ds_load_b128 per kt, zero repack
    v16bf afrag[8];
    {
        const uint4* ap = (const uint4*)sA;
#pragma unroll
        for (int kt = 0; kt < 8; ++kt) {
            uint4 q0 = ap[(kt * 32 + lane) * 2 + 0];
            uint4 q1 = ap[(kt * 32 + lane) * 2 + 1];
            v8u u;
            u[0] = q0.x; u[1] = q0.y; u[2] = q0.z; u[3] = q0.w;
            u[4] = q1.x; u[5] = q1.y; u[6] = q1.z; u[7] = q1.w;
            afrag[kt] = __builtin_bit_cast(v16bf, u);
        }
    }

    float pm[8], pv[8];
#pragma unroll
    for (int r = 0; r < 8; ++r) { pm[r] = 0.f; pv[r] = 0.f; }

    int l16 = lane & 15;
#pragma unroll
    for (int ntl = 0; ntl < 8; ++ntl) {
        int nt = wave * 8 + ntl;         // global n-tile 0..31
        v8f acc = {0.f, 0.f, 0.f, 0.f, 0.f, 0.f, 0.f, 0.f};
        const uint4* bp = wfrag + (size_t)((nt * 8) * 32 + lane) * 2;
#pragma unroll
        for (int kt = 0; kt < 8; ++kt) {
            uint4 q0 = bp[0], q1 = bp[1];
            v8u u;
            u[0] = q0.x; u[1] = q0.y; u[2] = q0.z; u[3] = q0.w;
            u[4] = q1.x; u[5] = q1.y; u[6] = q1.z; u[7] = q1.w;
            v16bf bfrag = __builtin_bit_cast(v16bf, u);
            acc = __builtin_amdgcn_wmma_f32_16x16x32_bf16(
                      false, afrag[kt], false, bfrag, (short)0, acc, false, false);
            bp += 64;                    // next fragment: 32 lanes * 2 uint4
        }
        // fused epilogue: + per-b bias, branch-free selu, dot with W21/W22.
        // D layout: lane -> column n, vgpr r -> row (r + 8*(lane>=16)).
        int n = nt * 16 + l16;
        float bia = bias[(size_t)b * HID_ + n];
        float c21 = w21[n], c22 = w22[n];
#pragma unroll
        for (int r = 0; r < 8; ++r) {
            float s = selu_f(acc[r] + bia);
            pm[r] += s * c21;
            pv[r] += s * c22;
        }
    }
    // reduce the 16 N-columns each half-wave holds
#pragma unroll
    for (int r = 0; r < 8; ++r) {
        float m = pm[r], v = pv[r];
#pragma unroll
        for (int d = 1; d < 16; d <<= 1) {
            m += __shfl_xor(m, d, 32);
            v += __shfl_xor(v, d, 32);
        }
        pm[r] = m; pv[r] = v;
    }
    if (l16 == 0) {                       // lanes 0 and 16: rows 0..7 / 8..15
        int mbase = (lane >> 4) * 8;
#pragma unroll
        for (int r = 0; r < 8; ++r) {
            atomicAdd(&redm[mbase + r], pm[r]);   // ds_add_f32, cross-wave combine
            atomicAdd(&redv[mbase + r], pv[r]);
        }
    }
    __syncthreads();
    if (tid < 16) {
        logm[row0 + tid] = redm[tid] + b21[0];
        logv[row0 + tid] = redv[tid] + b22[0];
    }
}

// ---------------------------------------------------------------------------
// K4: per-b dual logsumexp over S=256 + final outputs.
// out[0:512] = log_mu, out[512:1024] = log_lambda
// ---------------------------------------------------------------------------
__global__ __launch_bounds__(256) void k_finish(
        const float* __restrict__ logm, const float* __restrict__ logv,
        const float* __restrict__ w, const float* __restrict__ l,
        float* __restrict__ out) {
    __shared__ float sA[256], sB[256];
    int b = blockIdx.x, t = threadIdx.x;
    float lw = __logf(w[(size_t)b * S_ + t]);
    float am = logm[(size_t)b * S_ + t] + lw;
    float av = logv[(size_t)b * S_ + t] + 2.f * lw;
    sA[t] = am; sB[t] = av; __syncthreads();
    for (int st = 128; st > 0; st >>= 1) {
        if (t < st) { sA[t] = fmaxf(sA[t], sA[t + st]); sB[t] = fmaxf(sB[t], sB[t + st]); }
        __syncthreads();
    }
    float Mm = sA[0], Mv = sB[0];
    __syncthreads();
    sA[t] = __expf(am - Mm); sB[t] = __expf(av - Mv);
    __syncthreads();
    for (int st = 128; st > 0; st >>= 1) {
        if (t < st) { sA[t] += sA[t + st]; sB[t] += sB[t + st]; }
        __syncthreads();
    }
    if (t == 0) {
        float lmA = Mm + __logf(sA[0]);
        float lvA = Mv + __logf(sB[0]);
        float ll  = __logf(l[b]);
        out[b]      = ll - lmA;
        out[B_ + b] = ll - 3.f * lmA - lvA;
    }
}

extern "C" void kernel_launch(void* const* d_in, const int* in_sizes, int n_in,
                              void* d_out, int out_size, void* d_ws, size_t ws_size,
                              hipStream_t stream) {
    (void)in_sizes; (void)n_in; (void)out_size; (void)ws_size;
    const float* rho  = (const float*)d_in[0];
    const float* c    = (const float*)d_in[1];
    const float* w    = (const float*)d_in[2];
    const float* l    = (const float*)d_in[3];
    /* d_in[4] = roads : unused by the reference */
    const int*   lon  = (const int*)d_in[5];
    const int*   lat  = (const int*)d_in[6];
    const float* f2W1 = (const float*)d_in[7];
    const float* f2b1 = (const float*)d_in[8];
    const float* f2W2 = (const float*)d_in[9];
    const float* f2b2 = (const float*)d_in[10];
    const float* fW1  = (const float*)d_in[11];
    const float* fb1  = (const float*)d_in[12];
    const float* fW21 = (const float*)d_in[13];
    const float* fb21 = (const float*)d_in[14];
    const float* fW22 = (const float*)d_in[15];
    const float* fb22 = (const float*)d_in[16];

    char* ws = (char*)d_ws;
    uint4* wfrag = (uint4*)ws;                               // 256 KB bf16 fragments
    float* bias  = (float*)(ws + (256 << 10));               // 1 MB  (B x 512)
    float* logm  = (float*)(ws + (256 << 10) + (1 << 20));   // 512 KB
    float* logv  = logm + (size_t)B_ * S_;                   // 512 KB

    k_prep_w <<<256, 32, 0, stream>>>(fW1, wfrag);
    k_context<<<B_, 256, 0, stream>>>(c, lon, lat, f2W1, f2b1, f2W2, f2b2,
                                      fW1, fb1, bias);
    k_gemm   <<<(B_ * S_) / 16, 128, 0, stream>>>(rho, wfrag, bias,
                                                  fW21, fW22, fb21, fb22,
                                                  logm, logv);
    k_finish <<<B_, 256, 0, stream>>>(logm, logv, w, l, (float*)d_out);
}
